// PointNetSetAbstraction_50714973831836
// MI455X (gfx1250) — matrix operand order repaired and checked by reading, not compiled, not run on hardware
//
#include <hip/hip_runtime.h>

#define B_ 16
#define N_ 8192
#define S_ 1024   // NPOINT
#define K_ 32     // NSAMPLE
#define D_ 64
#define BTOT (B_*S_*K_)   // 524288 rows
#define EPSV 1e-5f

typedef __attribute__((ext_vector_type(16))) _Float16 v16h;
typedef __attribute__((ext_vector_type(8)))  float    v8f;

union Frag16 { v16h v; uint4 q[2]; };

// ---------------------------------------------------------------------------
// Farthest point sampling: one 1024-thread block per batch; points cached in
// registers (8/thread). Per-step argmax: wave32 shuffle butterfly + one small
// LDS stage (2 barriers/step). Winner coordinates are carried through the
// reduction so the critical path never touches global memory.
// ---------------------------------------------------------------------------
__global__ __launch_bounds__(1024)
void fps_kernel(const float* __restrict__ xyz, int* __restrict__ fpsIdx,
                float* __restrict__ newXyz, float* __restrict__ outXyzT)
{
  __shared__ float sval[32], svx[32], svy[32], svz[32];
  __shared__ int   sidx[32];
  __shared__ float cent[3];
  const int b = blockIdx.x;
  const int t = threadIdx.x;
  const int w = t >> 5, lane = t & 31;
  const float* xb = xyz + (size_t)b * 3 * N_;
  float px[8], py[8], pz[8], dist[8];
#pragma unroll
  for (int j = 0; j < 8; ++j) {
    int n = t + 1024 * j;
    px[j] = xb[n]; py[j] = xb[N_ + n]; pz[j] = xb[2 * N_ + n];
    dist[j] = 1e10f;
  }
  if (t == 0) {
    float cx = xb[0], cy = xb[N_], cz = xb[2 * N_];
    cent[0] = cx; cent[1] = cy; cent[2] = cz;
    fpsIdx[b * S_] = 0;
    float* nx = newXyz + (size_t)b * S_ * 3;
    nx[0] = cx; nx[1] = cy; nx[2] = cz;
    outXyzT[(size_t)b * 3 * S_ + 0 * S_] = cx;
    outXyzT[(size_t)b * 3 * S_ + 1 * S_] = cy;
    outXyzT[(size_t)b * 3 * S_ + 2 * S_] = cz;
  }
  __syncthreads();
  for (int s = 1; s < S_; ++s) {
    const float cx = cent[0], cy = cent[1], cz = cent[2];
    float best = -1.f; int bi = 0; float bx = 0.f, by = 0.f, bz = 0.f;
#pragma unroll
    for (int j = 0; j < 8; ++j) {
      int n = t + 1024 * j;
      float dx = px[j] - cx, dy = py[j] - cy, dz = pz[j] - cz;
      float d = dx * dx + dy * dy + dz * dz;
      dist[j] = fminf(dist[j], d);
      if (dist[j] > best || (dist[j] == best && n < bi)) {
        best = dist[j]; bi = n; bx = px[j]; by = py[j]; bz = pz[j];
      }
    }
#pragma unroll
    for (int off = 16; off > 0; off >>= 1) {
      float ov = __shfl_xor(best, off, 32);
      int   oi = __shfl_xor(bi,   off, 32);
      float ox = __shfl_xor(bx,   off, 32);
      float oy = __shfl_xor(by,   off, 32);
      float oz = __shfl_xor(bz,   off, 32);
      if (ov > best || (ov == best && oi < bi)) { best = ov; bi = oi; bx = ox; by = oy; bz = oz; }
    }
    if (lane == 0) { sval[w] = best; sidx[w] = bi; svx[w] = bx; svy[w] = by; svz[w] = bz; }
    __syncthreads();
    if (t < 32) {
      best = sval[t]; bi = sidx[t]; bx = svx[t]; by = svy[t]; bz = svz[t];
#pragma unroll
      for (int off = 16; off > 0; off >>= 1) {
        float ov = __shfl_xor(best, off, 32);
        int   oi = __shfl_xor(bi,   off, 32);
        float ox = __shfl_xor(bx,   off, 32);
        float oy = __shfl_xor(by,   off, 32);
        float oz = __shfl_xor(bz,   off, 32);
        if (ov > best || (ov == best && oi < bi)) { best = ov; bi = oi; bx = ox; by = oy; bz = oz; }
      }
      if (t == 0) {
        cent[0] = bx; cent[1] = by; cent[2] = bz;
        fpsIdx[b * S_ + s] = bi;
        float* nx = newXyz + ((size_t)b * S_ + s) * 3;
        nx[0] = bx; nx[1] = by; nx[2] = bz;
        outXyzT[(size_t)b * 3 * S_ + 0 * S_ + s] = bx;
        outXyzT[(size_t)b * 3 * S_ + 1 * S_ + s] = by;
        outXyzT[(size_t)b * 3 * S_ + 2 * S_ + s] = bz;
      }
    }
    __syncthreads();
  }
}

// ---------------------------------------------------------------------------
// KNN: one block (8 waves) per query; all 8192 distances in LDS, 32 selection
// rounds; argmin via wave shuffle butterfly + 8-entry LDS stage (2 barriers).
// ---------------------------------------------------------------------------
__global__ __launch_bounds__(256)
void knn_kernel(const float* __restrict__ xyz, const float* __restrict__ newXyz,
                int* __restrict__ knnIdx)
{
  __shared__ float sd[N_];
  __shared__ float rv[8];
  __shared__ int   ri[8];
  const int q = blockIdx.x;          // b*S + s
  const int b = q >> 10;
  const int t = threadIdx.x;
  const int w = t >> 5, lane = t & 31;
  const float* xb = xyz + (size_t)b * 3 * N_;
  const float* nx = newXyz + (size_t)q * 3;
  const float qx = nx[0], qy = nx[1], qz = nx[2];
#pragma unroll
  for (int j = 0; j < N_ / 256; ++j) {
    int n = t + 256 * j;
    float dx = xb[n] - qx, dy = xb[N_ + n] - qy, dz = xb[2 * N_ + n] - qz;
    sd[n] = dx * dx + dy * dy + dz * dz;
  }
  __syncthreads();
  for (int r = 0; r < K_; ++r) {
    float best = 3.0e38f; int bi = t;
#pragma unroll
    for (int j = 0; j < N_ / 256; ++j) {
      int n = t + 256 * j;
      float d = sd[n];
      if (d < best || (d == best && n < bi)) { best = d; bi = n; }
    }
#pragma unroll
    for (int off = 16; off > 0; off >>= 1) {
      float ov = __shfl_xor(best, off, 32);
      int   oi = __shfl_xor(bi,   off, 32);
      if (ov < best || (ov == best && oi < bi)) { best = ov; bi = oi; }
    }
    if (lane == 0) { rv[w] = best; ri[w] = bi; }
    __syncthreads();
    if (t < 8) {
      best = rv[t]; bi = ri[t];
#pragma unroll
      for (int off = 4; off > 0; off >>= 1) {
        float ov = __shfl_xor(best, off, 32);
        int   oi = __shfl_xor(bi,   off, 32);
        if (ov < best || (ov == best && oi < bi)) { best = ov; bi = oi; }
      }
      if (t == 0) { knnIdx[(size_t)q * K_ + r] = bi; sd[bi] = 3.4e38f; }
    }
    __syncthreads();
  }
}

// ---------------------------------------------------------------------------
// Layer 0: fused gather (xyz-rel + point features, K padded 67->96) + WMMA GEMM
// y0 = x0 @ W0^T + b0, f16 out, per-channel sum/sumsq accumulated for BN.
// knn-index tile staged into LDS with GLOBAL_LOAD_ASYNC_TO_LDS_B128 (ASYNCcnt).
// ---------------------------------------------------------------------------
__global__ __launch_bounds__(256)
void gemm0_kernel(const float* __restrict__ xyz, const float* __restrict__ points,
                  const float* __restrict__ newXyz, const int* __restrict__ knnIdx,
                  const float* __restrict__ W0, const float* __restrict__ b0,
                  _Float16* __restrict__ y0,
                  float* __restrict__ gsum, float* __restrict__ gsq)
{
  __shared__ __align__(16) _Float16 sA[128][96];
  __shared__ __align__(16) _Float16 sW[64][96];
  __shared__ __align__(16) int spidx[128];
  __shared__ float lsum[64], lsq[64];
  const int t = threadIdx.x;
  const int rowBase = blockIdx.x * 128;
  const int b = rowBase >> 15;            // 32768 rows per batch
  // Async copy of the 128 contiguous knn indices for this tile into LDS.
  if (t < 32) {
    unsigned int ldsOff  = (unsigned int)(uintptr_t)(&spidx[0]) + 16u * (unsigned int)t;
    unsigned int byteOff = 16u * (unsigned int)t;
    const int* src = knnIdx + rowBase;
    asm volatile("global_load_async_to_lds_b128 %0, %1, %2"
                 :: "v"(ldsOff), "v"(byteOff), "s"(src) : "memory");
  }
  if (t < 64) { lsum[t] = 0.f; lsq[t] = 0.f; }
  for (int i = t; i < 64 * 96; i += 256) {
    int o = i / 96, c = i % 96;
    sW[o][c] = (c < 67) ? (_Float16)W0[o * 67 + c] : (_Float16)0.f;
  }
  asm volatile("s_wait_asynccnt 0x0" ::: "memory");
  __syncthreads();
  {
    const int rr = t & 127, hh = t >> 7;       // 2 threads per row
    const int r = rowBase + rr;
    const int s = (r >> 5) & 1023;
    const int gi = b * S_ + s;
    const int pidx = spidx[rr];
    const float* xb = xyz + (size_t)b * 3 * N_;
    const float* pb = points + (size_t)b * D_ * N_;
    const float* nx = newXyz + (size_t)gi * 3;
    const int c0 = hh * 48, c1 = c0 + 48;
    for (int c = c0; c < c1; ++c) {
      float v;
      if (c < 3)       v = xb[c * N_ + pidx] - nx[c];
      else if (c < 67) v = pb[(c - 3) * N_ + pidx];
      else             v = 0.f;
      sA[rr][c] = (_Float16)v;
    }
  }
  __syncthreads();
  const int w = t >> 5, l = t & 31;
  const int h = l >> 4, ln = l & 15;
  v8f acc[4];
#pragma unroll
  for (int nt = 0; nt < 4; ++nt) {
    float bv = b0[nt * 16 + ln];
#pragma unroll
    for (int r = 0; r < 8; ++r) acc[nt][r] = bv;
  }
#pragma unroll
  for (int kt = 0; kt < 3; ++kt) {
    Frag16 fa;
    const int arow = 16 * w + ln;
    fa.q[0] = *(const uint4*)&sA[arow][kt * 32 + 8 * h];
    fa.q[1] = *(const uint4*)&sA[arow][kt * 32 + 16 + 8 * h];
#pragma unroll
    for (int nt = 0; nt < 4; ++nt) {
      Frag16 fb;
      const int col = nt * 16 + ln;
      fb.q[0] = *(const uint4*)&sW[col][kt * 32 + 8 * h];
      fb.q[1] = *(const uint4*)&sW[col][kt * 32 + 16 + 8 * h];
      acc[nt] = __builtin_amdgcn_wmma_f32_16x16x32_f16(false, fa.v, false, fb.v,
                                                       (short)0, acc[nt], false, false);
    }
  }
#pragma unroll
  for (int nt = 0; nt < 4; ++nt) {
    const int col = nt * 16 + ln;
    float s1 = 0.f, s2 = 0.f;
#pragma unroll
    for (int r = 0; r < 8; ++r) {
      float v = acc[nt][r];
      int grow = rowBase + 16 * w + 8 * h + r;
      y0[(size_t)grow * 64 + col] = (_Float16)v;
      s1 += v; s2 += v * v;
    }
    atomicAdd(&lsum[col], s1);
    atomicAdd(&lsq[col], s2);
  }
  __syncthreads();
  if (t < 64) { atomicAdd(&gsum[t], lsum[t]); atomicAdd(&gsq[t], lsq[t]); }
}

// ---------------------------------------------------------------------------
// BN stat finalize: scale = g*rsqrt(var+eps), shift = be - mu*scale
// ---------------------------------------------------------------------------
__global__ void bnstats_kernel(const float* __restrict__ sum, const float* __restrict__ sq,
                               const float* __restrict__ g, const float* __restrict__ be,
                               float* __restrict__ scale, float* __restrict__ shift, int nch)
{
  int c = threadIdx.x;
  if (c < nch) {
    const float inv = 1.f / (float)BTOT;
    float mu = sum[c] * inv;
    float var = sq[c] * inv - mu * mu;
    float sc = g[c] * rsqrtf(var + EPSV);
    scale[c] = sc;
    shift[c] = be[c] - mu * sc;
  }
}

// ---------------------------------------------------------------------------
// Layers 1/2: BN+ReLU of previous layer fused into A-fragment staging, then
// WMMA GEMM (K=64, NOUT templated), f16 out + BN stat accumulation.
// ---------------------------------------------------------------------------
template<int NOUT>
__global__ __launch_bounds__(256)
void gemm_bn_kernel(const _Float16* __restrict__ yIn,
                    const float* __restrict__ scale, const float* __restrict__ shift,
                    const float* __restrict__ W, const float* __restrict__ bias,
                    _Float16* __restrict__ yOut,
                    float* __restrict__ gsum, float* __restrict__ gsq)
{
  __shared__ __align__(16) _Float16 sA[128][64];
  __shared__ __align__(16) _Float16 sW[NOUT][64];
  __shared__ float lsum[NOUT], lsq[NOUT];
  const int t = threadIdx.x;
  const int rowBase = blockIdx.x * 128;
  // Get the activation tile in flight early (global_prefetch_b8).
  if (t < 128) __builtin_prefetch(&yIn[(size_t)(rowBase + t) * 64], 0, 0);
  for (int i = t; i < NOUT; i += 256) { lsum[i] = 0.f; lsq[i] = 0.f; }
  for (int i = t; i < NOUT * 64; i += 256) {
    int o = i >> 6, c = i & 63;
    sW[o][c] = (_Float16)W[i];
  }
  for (int i = t; i < 128 * 64; i += 256) {
    int rr = i >> 6, c = i & 63;
    float v = (float)yIn[((size_t)(rowBase + rr) << 6) + c];
    v = fmaxf(v * scale[c] + shift[c], 0.f);
    sA[rr][c] = (_Float16)v;
  }
  __syncthreads();
  const int w = t >> 5, l = t & 31;
  const int h = l >> 4, ln = l & 15;
  constexpr int NT = NOUT / 16;
  v8f acc[NT];
#pragma unroll
  for (int nt = 0; nt < NT; ++nt) {
    float bv = bias[nt * 16 + ln];
#pragma unroll
    for (int r = 0; r < 8; ++r) acc[nt][r] = bv;
  }
#pragma unroll
  for (int kt = 0; kt < 2; ++kt) {
    Frag16 fa;
    const int arow = 16 * w + ln;
    fa.q[0] = *(const uint4*)&sA[arow][kt * 32 + 8 * h];
    fa.q[1] = *(const uint4*)&sA[arow][kt * 32 + 16 + 8 * h];
#pragma unroll
    for (int nt = 0; nt < NT; ++nt) {
      Frag16 fb;
      const int col = nt * 16 + ln;
      fb.q[0] = *(const uint4*)&sW[col][kt * 32 + 8 * h];
      fb.q[1] = *(const uint4*)&sW[col][kt * 32 + 16 + 8 * h];
      acc[nt] = __builtin_amdgcn_wmma_f32_16x16x32_f16(false, fa.v, false, fb.v,
                                                       (short)0, acc[nt], false, false);
    }
  }
#pragma unroll
  for (int nt = 0; nt < NT; ++nt) {
    const int col = nt * 16 + ln;
    float s1 = 0.f, s2 = 0.f;
#pragma unroll
    for (int r = 0; r < 8; ++r) {
      float v = acc[nt][r];
      int grow = rowBase + 16 * w + 8 * h + r;
      yOut[(size_t)grow * NOUT + col] = (_Float16)v;
      s1 += v; s2 += v * v;
    }
    atomicAdd(&lsum[col], s1);
    atomicAdd(&lsq[col], s2);
  }
  __syncthreads();
  for (int i = t; i < NOUT; i += 256) { atomicAdd(&gsum[i], lsum[i]); atomicAdd(&gsq[i], lsq[i]); }
}

// ---------------------------------------------------------------------------
// Final BN+ReLU + max over nsample, written transposed [B,128,S].
// ---------------------------------------------------------------------------
__global__ __launch_bounds__(128)
void maxpool_kernel(const _Float16* __restrict__ y2,
                    const float* __restrict__ scale, const float* __restrict__ shift,
                    float* __restrict__ outPts)
{
  const int q = blockIdx.x;            // b*S + s
  const int c = threadIdx.x;
  const int b = q >> 10, s = q & 1023;
  const float sc = scale[c], sh = shift[c];
  float m = 0.f;                       // ReLU outputs are >= 0
#pragma unroll
  for (int k = 0; k < K_; ++k) {
    float v = (float)y2[(((size_t)q * K_) + k) * 128 + c];
    m = fmaxf(m, fmaxf(v * sc + sh, 0.f));
  }
  outPts[((size_t)b * 128 + c) * S_ + s] = m;
}

// ---------------------------------------------------------------------------
extern "C" void kernel_launch(void* const* d_in, const int* in_sizes, int n_in,
                              void* d_out, int out_size, void* d_ws, size_t ws_size,
                              hipStream_t stream)
{
  (void)in_sizes; (void)n_in; (void)out_size; (void)ws_size;
  const float* xyz    = (const float*)d_in[0];
  const float* points = (const float*)d_in[1];
  const float* W0  = (const float*)d_in[2];
  const float* b0  = (const float*)d_in[3];
  const float* g0  = (const float*)d_in[4];
  const float* be0 = (const float*)d_in[5];
  const float* W1  = (const float*)d_in[6];
  const float* b1  = (const float*)d_in[7];
  const float* g1  = (const float*)d_in[8];
  const float* be1 = (const float*)d_in[9];
  const float* W2  = (const float*)d_in[10];
  const float* b2  = (const float*)d_in[11];
  const float* g2  = (const float*)d_in[12];
  const float* be2 = (const float*)d_in[13];

  char* ws = (char*)d_ws;
  int*      fpsIdx = (int*)(ws + 0);                        //  64 KB
  float*    newXyz = (float*)(ws + 65536);                  // 192 KB
  int*      knnIdx = (int*)(ws + 262144);                   //   2 MB
  float*    stats  = (float*)(ws + 2359296);                //   4 KB
  _Float16* y0     = (_Float16*)(ws + 2363392);             //  64 MB
  _Float16* y1     = (_Float16*)(ws + 2363392 + 67108864);  //  64 MB
  _Float16* y2     = (_Float16*)(ws + 2363392 + 2 * 67108864); // 128 MB

  float* sum0 = stats +   0; float* sq0 = stats +  64;
  float* sum1 = stats + 128; float* sq1 = stats + 192;
  float* sum2 = stats + 256; float* sq2 = stats + 384;
  float* sc0  = stats + 512; float* sh0 = stats + 576;
  float* sc1  = stats + 640; float* sh1 = stats + 704;
  float* sc2  = stats + 768; float* sh2 = stats + 896;

  float* outXyzT = (float*)d_out;                           // [B,3,S]
  float* outPts  = (float*)d_out + (size_t)B_ * 3 * S_;     // [B,128,S]

  hipMemsetAsync(stats, 0, 2048, stream);   // zero sum/sq accumulators

  fps_kernel<<<B_, 1024, 0, stream>>>(xyz, fpsIdx, newXyz, outXyzT);
  knn_kernel<<<B_ * S_, 256, 0, stream>>>(xyz, newXyz, knnIdx);
  gemm0_kernel<<<BTOT / 128, 256, 0, stream>>>(xyz, points, newXyz, knnIdx, W0, b0,
                                               y0, sum0, sq0);
  bnstats_kernel<<<1, 128, 0, stream>>>(sum0, sq0, g0, be0, sc0, sh0, 64);
  gemm_bn_kernel<64><<<BTOT / 128, 256, 0, stream>>>(y0, sc0, sh0, W1, b1, y1, sum1, sq1);
  bnstats_kernel<<<1, 128, 0, stream>>>(sum1, sq1, g1, be1, sc1, sh1, 64);
  gemm_bn_kernel<128><<<BTOT / 128, 256, 0, stream>>>(y1, sc1, sh1, W2, b2, y2, sum2, sq2);
  bnstats_kernel<<<1, 128, 0, stream>>>(sum2, sq2, g2, be2, sc2, sh2, 128);
  maxpool_kernel<<<B_ * S_, 128, 0, stream>>>(y2, sc2, sh2, outPts);
}
